// GAT_8083128451589
// MI455X (gfx1250) — matrix operand order repaired and checked by reading, not compiled, not run on hardware
//
#include <hip/hip_runtime.h>

typedef __attribute__((ext_vector_type(16))) _Float16 v16h;
typedef __attribute__((ext_vector_type(8)))  float    v8f;
typedef __attribute__((ext_vector_type(4)))  int      v4i;

#define GAT_NEG_SLOPE 0.2f
// order-preserving map of -inf: bits(-inf)=0xFF800000 -> ~bits = 0x007FFFFF
#define NEG_INF_ORD 0x007FFFFFu

#define AS1 __attribute__((address_space(1)))
#define AS3 __attribute__((address_space(3)))

// ---------------- small helpers ----------------

__device__ __forceinline__ unsigned gat_f2ord(float f) {
  unsigned u = __float_as_uint(f);
  return (u & 0x80000000u) ? ~u : (u | 0x80000000u);
}
__device__ __forceinline__ float gat_ord2f(unsigned o) {
  unsigned u = (o & 0x80000000u) ? (o ^ 0x80000000u) : ~o;
  return __uint_as_float(u);
}
__device__ __forceinline__ void gat_edge_sd(const int* __restrict__ ei, int E,
                                            int e, int& src, int& dst) {
  if (e < E) { src = ei[e]; dst = ei[E + e]; }
  else       { src = e - E; dst = src; }      // self-loop
}

__global__ void k_fill_f32(float* __restrict__ p, float v, int n) {
  int i = blockIdx.x * blockDim.x + threadIdx.x;
  if (i < n) p[i] = v;
}
__global__ void k_fill_u32(unsigned* __restrict__ p, unsigned v, int n) {
  int i = blockIdx.x * blockDim.x + threadIdx.x;
  if (i < n) p[i] = v;
}

// ---------------- WMMA GEMM: C[M,N] = A[M,K] @ B[K,N] (f32 in/out, f16 MACs) ----
// One block per 16-row M stripe; blockDim.x = 32 * (N/16) waves, wave w owns
// N-tile w. A stripe (16 x K f32) staged once into LDS (async-to-LDS when the
// toolchain exposes it), shared by all waves. K compile-time -> k-loop fully
// unrolled (K/32 v_wmma per wave). Requires M%16==0, N%16==0, K%32==0, K%4==0.
template <int K>
__global__ void k_gemm_wmma_f16(const float* __restrict__ A,
                                const float* __restrict__ B,
                                float* __restrict__ C, int N) {
  __shared__ float Atile[16 * K];
  const int tid = threadIdx.x;
  const int m0  = blockIdx.x * 16;

  // ---- stage A stripe into LDS (16-byte granules) ----
  {
    const int stride = (int)blockDim.x * 4;
#if defined(__gfx1250__) && __has_builtin(__builtin_amdgcn_global_load_async_to_lds_b128)
    for (int i = tid * 4; i < 16 * K; i += stride) {
      int row = i / K, col = i - row * K;
      void* gp = (void*)(A + (size_t)(m0 + row) * K + col);
      void* lp = (void*)(Atile + i);
      __builtin_amdgcn_global_load_async_to_lds_b128(
          (AS1 v4i*)gp, (AS3 v4i*)lp, /*offset=*/0, /*cpol=*/0);
    }
#if __has_builtin(__builtin_amdgcn_s_wait_asynccnt)
    __builtin_amdgcn_s_wait_asynccnt(0);
#else
    asm volatile("s_wait_asynccnt 0" ::: "memory");
#endif
#else
    for (int i = tid * 4; i < 16 * K; i += stride) {
      int row = i / K, col = i - row * K;
      *(float4*)(Atile + i) = *(const float4*)(A + (size_t)(m0 + row) * K + col);
    }
#endif
  }
  __syncthreads();

  const int lane = tid & 31;
  const int wave = tid >> 5;       // N-tile index
  const int r    = lane & 15;      // M-in-tile for A, N-in-tile for B/C
  const int hl   = lane >> 4;      // half selector per ISA layout
  const int col0 = wave * 16;

  v8f c = {};
  const float* arow = Atile + r * K + hl * 8;               // LDS
  const float* bcol = B + (size_t)(hl * 8) * N + (col0 + r);

#pragma unroll
  for (int k = 0; k < K; k += 32) {
    // A 16x32 f16 frag: slot i -> K = k + hl*8 + (i&7) + (i>=8)*16
    float af[16];
    *(float4*)(af + 0)  = *(const float4*)(arow + k);
    *(float4*)(af + 4)  = *(const float4*)(arow + k + 4);
    *(float4*)(af + 8)  = *(const float4*)(arow + k + 16);
    *(float4*)(af + 12) = *(const float4*)(arow + k + 20);
    v16h a, b;
#pragma unroll
    for (int i = 0; i < 16; ++i) a[i] = (_Float16)af[i];
    // B 32x16 f16 frag: slot i -> K = k + hl*8 + (i&7) + (i>=8)*16
#pragma unroll
    for (int i = 0; i < 16; ++i) b[i] = (_Float16)bcol[(size_t)(k + i + (i >= 8) * 8) * N];
    c = __builtin_amdgcn_wmma_f32_16x16x32_f16(false, a, false, b,
                                               (short)0, c, false, false);
  }
  // C/D: slot i -> M = i + hl*8, lane&15 -> N
  float* crow = C + (size_t)(m0 + hl * 8) * N + (col0 + r);
#pragma unroll
  for (int i = 0; i < 8; ++i) crow[(size_t)i * N] = c[i];
}

// ---------------- per-node attention dot products --------------------------
__global__ void k_attn_terms(const float* __restrict__ h,
                             const float* __restrict__ asrc,
                             const float* __restrict__ adst,
                             float* __restrict__ out_s,
                             float* __restrict__ out_d,
                             int n, int H, int C) {
  int idx = blockIdx.x * blockDim.x + threadIdx.x;
  if (idx >= n * H) return;
  int node = idx / H, hh = idx - node * H;
  const float* hp = h + (size_t)node * H * C + (size_t)hh * C;
  const float* sp = asrc + hh * C;
  const float* dp = adst + hh * C;
  float s = 0.f, d = 0.f;
  for (int cc = 0; cc < C; ++cc) { float v = hp[cc]; s += v * sp[cc]; d += v * dp[cc]; }
  out_s[idx] = s;
  out_d[idx] = d;
}

// ---------------- edge pass 1: leaky-relu score + segment max --------------
__global__ void k_edge_max(const int* __restrict__ ei, int E, int n,
                           const float* __restrict__ as_, const float* __restrict__ ad_,
                           float* __restrict__ escore, unsigned* __restrict__ maxb,
                           int H) {
  int idx = blockIdx.x * blockDim.x + threadIdx.x;
  int total = (E + n) * H;
  if (idx >= total) return;
  int e = idx / H, hh = idx - e * H;
  int src, dst;
  gat_edge_sd(ei, E, e, src, dst);
  float x = as_[src * H + hh] + ad_[dst * H + hh];
  float sc = (x > 0.f) ? x : GAT_NEG_SLOPE * x;
  escore[idx] = sc;
  atomicMax(&maxb[dst * H + hh], gat_f2ord(sc));
}

// ---------------- edge pass 2: exp(score - max) + segment sum --------------
__global__ void k_edge_expsum(const int* __restrict__ ei, int E, int n,
                              float* __restrict__ escore,
                              const unsigned* __restrict__ maxb,
                              float* __restrict__ sums, int H) {
  int idx = blockIdx.x * blockDim.x + threadIdx.x;
  int total = (E + n) * H;
  if (idx >= total) return;
  int e = idx / H, hh = idx - e * H;
  int src, dst;
  gat_edge_sd(ei, E, e, src, dst);
  float ex = __expf(escore[idx] - gat_ord2f(maxb[dst * H + hh]));
  escore[idx] = ex;
  atomicAdd(&sums[dst * H + hh], ex);
}

// ---------------- edge pass 3: alpha-weighted scatter of h[src] ------------
// One wave per edge; channels strided across lanes.
__global__ void k_edge_scatter(const int* __restrict__ ei, int E, int n,
                               const float* __restrict__ escore,
                               const float* __restrict__ sums,
                               const float* __restrict__ hlin,
                               float* __restrict__ acc, int H, int C) {
  int t = blockIdx.x * blockDim.x + threadIdx.x;
  int lane = t & 31;
  int e = t >> 5;
  if (e >= E + n) return;
  int src, dst;
  gat_edge_sd(ei, E, e, src, dst);
  const int HC = H * C;
  const float* hs = hlin + (size_t)src * HC;
  float* ad       = acc  + (size_t)dst * HC;
  for (int ch = lane; ch < HC; ch += 32) {
    int hh = ch / C;
    float alpha = escore[e * H + hh] / sums[dst * H + hh];
    atomicAdd(&ad[ch], hs[ch] * alpha);
  }
}

// ---------------- finalize: +bias then ELU (layer 1) -----------------------
__global__ void k_finalize_elu(float* __restrict__ acc,
                               const float* __restrict__ bias, int n, int HC) {
  int idx = blockIdx.x * blockDim.x + threadIdx.x;
  if (idx >= n * HC) return;
  float v = acc[idx] + bias[idx % HC];
  acc[idx] = (v > 0.f) ? v : (expm1f(v));
}

// ---------------- finalize: +bias only (layer 2, heads=1 mean==identity) ---
__global__ void k_finalize_bias(float* __restrict__ out,
                                const float* __restrict__ bias, int n, int C) {
  int idx = blockIdx.x * blockDim.x + threadIdx.x;
  if (idx >= n * C) return;
  out[idx] += bias[idx % C];
}

// ---------------------------------------------------------------------------

static inline int gat_cdiv(int a, int b) { return (a + b - 1) / b; }

extern "C" void kernel_launch(void* const* d_in, const int* in_sizes, int n_in,
                              void* d_out, int out_size, void* d_ws, size_t ws_size,
                              hipStream_t stream) {
  const float* x        = (const float*)d_in[0];
  const int*   ei       = (const int*)  d_in[1];   // [2,E] row-major
  const float* W1       = (const float*)d_in[2];   // [128,128]
  const float* att_src1 = (const float*)d_in[3];   // [2,64]
  const float* att_dst1 = (const float*)d_in[4];   // [2,64]
  const float* bias1    = (const float*)d_in[5];   // [128]
  const float* W2       = (const float*)d_in[6];   // [128,32]
  const float* att_src2 = (const float*)d_in[7];   // [1,32]
  const float* att_dst2 = (const float*)d_in[8];   // [1,32]
  const float* bias2    = (const float*)d_in[9];   // [32]

  const int NN = 50000, EE = 800000;
  const int ET = EE + NN;            // edges incl. self-loops
  const int IN = 128, HID = 64, HEADS = 2, OUT = 32;
  const int HC1 = HEADS * HID;       // 128
  (void)in_sizes; (void)n_in; (void)out_size; (void)ws_size; (void)IN;

  // ---- carve workspace ----
  char* base = (char*)d_ws;
  size_t off = 0;
  auto take = [&](size_t bytes) -> char* {
    char* p = base + off;
    off = (off + bytes + 255) & ~(size_t)255;
    return p;
  };
  float*    h1lin   = (float*)take((size_t)NN * HC1 * 4);   // X@W1
  float*    acc1    = (float*)take((size_t)NN * HC1 * 4);   // layer-1 agg / layer-2 input
  float*    as1     = (float*)take((size_t)NN * HEADS * 4);
  float*    ad1     = (float*)take((size_t)NN * HEADS * 4);
  float*    sum1    = (float*)take((size_t)NN * HEADS * 4);
  unsigned* max1    = (unsigned*)take((size_t)NN * HEADS * 4);
  float*    esc1    = (float*)take((size_t)ET * HEADS * 4);
  float*    h2lin   = (float*)take((size_t)NN * OUT * 4);   // H@W2
  float*    as2     = (float*)take((size_t)NN * 4);
  float*    ad2     = (float*)take((size_t)NN * 4);
  float*    sum2    = (float*)take((size_t)NN * 4);
  unsigned* max2    = (unsigned*)take((size_t)NN * 4);
  float*    esc2    = (float*)take((size_t)ET * 4);
  float*    out     = (float*)d_out;                        // [NN, 32]

  const int TB = 256;

  // ================= Layer 1 =================
  // h1lin = x @ W1 : N=128 -> 8 waves/block (256 threads), K=128 unrolled
  k_gemm_wmma_f16<128><<<NN / 16, 256, 0, stream>>>(x, W1, h1lin, HC1);
  // attention dot products
  k_attn_terms<<<gat_cdiv(NN * HEADS, TB), TB, 0, stream>>>(
      h1lin, att_src1, att_dst1, as1, ad1, NN, HEADS, HID);
  // init reductions
  k_fill_u32<<<gat_cdiv(NN * HEADS, TB), TB, 0, stream>>>(max1, NEG_INF_ORD, NN * HEADS);
  k_fill_f32<<<gat_cdiv(NN * HEADS, TB), TB, 0, stream>>>(sum1, 0.f, NN * HEADS);
  k_fill_f32<<<gat_cdiv(NN * HC1, TB), TB, 0, stream>>>(acc1, 0.f, NN * HC1);
  // softmax + aggregate over edges
  k_edge_max<<<gat_cdiv(ET * HEADS, TB), TB, 0, stream>>>(
      ei, EE, NN, as1, ad1, esc1, max1, HEADS);
  k_edge_expsum<<<gat_cdiv(ET * HEADS, TB), TB, 0, stream>>>(
      ei, EE, NN, esc1, max1, sum1, HEADS);
  k_edge_scatter<<<gat_cdiv(ET * 32, TB), TB, 0, stream>>>(
      ei, EE, NN, esc1, sum1, h1lin, acc1, HEADS, HID);
  // +bias1 then ELU -> acc1 becomes layer-2 input
  k_finalize_elu<<<gat_cdiv(NN * HC1, TB), TB, 0, stream>>>(acc1, bias1, NN, HC1);

  // ================= Layer 2 =================
  // h2lin = elu(h1) @ W2 : N=32 -> 2 waves/block (64 threads), K=128 unrolled
  k_gemm_wmma_f16<128><<<NN / 16, 64, 0, stream>>>(acc1, W2, h2lin, OUT);
  k_attn_terms<<<gat_cdiv(NN, TB), TB, 0, stream>>>(
      h2lin, att_src2, att_dst2, as2, ad2, NN, 1, OUT);
  k_fill_u32<<<gat_cdiv(NN, TB), TB, 0, stream>>>(max2, NEG_INF_ORD, NN);
  k_fill_f32<<<gat_cdiv(NN, TB), TB, 0, stream>>>(sum2, 0.f, NN);
  k_fill_f32<<<gat_cdiv(NN * OUT, TB), TB, 0, stream>>>(out, 0.f, NN * OUT);
  k_edge_max<<<gat_cdiv(ET, TB), TB, 0, stream>>>(
      ei, EE, NN, as2, ad2, esc2, max2, 1);
  k_edge_expsum<<<gat_cdiv(ET, TB), TB, 0, stream>>>(
      ei, EE, NN, esc2, max2, sum2, 1);
  k_edge_scatter<<<gat_cdiv(ET * 32, TB), TB, 0, stream>>>(
      ei, EE, NN, esc2, sum2, h2lin, out, 1, OUT);
  k_finalize_bias<<<gat_cdiv(NN * OUT, TB), TB, 0, stream>>>(out, bias2, NN, OUT);
}
